// FreqFusionBlock_111669149832
// MI455X (gfx1250) — matrix operand order repaired and checked by reading, not compiled, not run on hardware
//
#include <hip/hip_runtime.h>
#include <hip/hip_bf16.h>

typedef _Float16 v16h __attribute__((ext_vector_type(16)));
typedef _Float16 v8h  __attribute__((ext_vector_type(8)));
typedef float    v8f  __attribute__((ext_vector_type(8)));
typedef int      v4i  __attribute__((ext_vector_type(4)));

constexpr int NB   = 4;       // batch
constexpr int NC   = 256;     // channels
constexpr int NH   = 96;
constexpr int NW   = 96;
constexpr int NHW  = NH * NW;           // 9216
constexpr int NPIX = NB * NH * NW;      // 36864 pixels
constexpr int TILES_X = NW / 16;        // 6
constexpr int MTILES  = NPIX / 16;      // 2304

constexpr int NGROUP_CO = 64;           // couts per block
constexpr int WAVES_PB  = 8;            // waves per block (256 threads)
constexpr int NSUB      = 4;            // 16-wide n-subtiles per wave

// Async global->LDS staging (gfx1250 GLOBAL_LOAD_ASYNC_TO_LDS_B128), guarded so
// the file still compiles on toolchains without the builtin.
#if defined(__has_builtin)
#  if __has_builtin(__builtin_amdgcn_global_load_async_to_lds_b128) && \
      __has_builtin(__builtin_amdgcn_s_wait_asynccnt)
#    define USE_ASYNC_LDS 1
#  endif
#endif
#ifndef USE_ASYNC_LDS
#  define USE_ASYNC_LDS 0
#endif

__device__ inline v16h zero_v16h() {
  v16h v;
#pragma unroll
  for (int i = 0; i < 16; ++i) v[i] = (_Float16)0.0f;
  return v;
}

// ---------------------------------------------------------------------------
// NCHW f32 -> NHWC f16 conversion
// ---------------------------------------------------------------------------
__global__ void k_nchw2nhwc_h(const float* __restrict__ in, _Float16* __restrict__ out, int C) {
  long tid = (long)blockIdx.x * blockDim.x + threadIdx.x;
  if (tid >= (long)NPIX * C) return;
  int c = (int)(tid % C);
  int p = (int)(tid / C);
  int b = p / NHW, rem = p % NHW;
  int y = rem / NW, x = rem % NW;
  out[tid] = (_Float16)in[(((long)b * C + c) * NH + y) * NW + x];
}

// ---------------------------------------------------------------------------
// Weight repack: OIHW f32 -> [co][tap*CinP + ci] f16, zero-padded ci>=Cin
// ---------------------------------------------------------------------------
__global__ void k_repack_w(const float* __restrict__ w, _Float16* __restrict__ wk,
                           int Cout, int Cin, int CinP, int taps) {
  long tid = (long)blockIdx.x * blockDim.x + threadIdx.x;
  long total = (long)Cout * taps * CinP;
  if (tid >= total) return;
  int ci  = (int)(tid % CinP);
  long r  = tid / CinP;
  int t   = (int)(r % taps);
  int co  = (int)(r / taps);
  float v = 0.0f;
  if (ci < Cin) v = w[((long)co * Cin + ci) * taps + t];
  wk[(long)co * ((long)taps * CinP) + (long)t * CinP + ci] = (_Float16)v;
}

// ---------------------------------------------------------------------------
// Depthwise 3x3 (pad=1) on NCHW f32 input -> NHWC f16
// ---------------------------------------------------------------------------
__global__ void k_dw3x3(const float* __restrict__ in, const float* __restrict__ w,
                        const float* __restrict__ bias, _Float16* __restrict__ out) {
  long tid = (long)blockIdx.x * blockDim.x + threadIdx.x;
  if (tid >= (long)NPIX * NC) return;
  int c = (int)(tid % NC);
  int p = (int)(tid / NC);
  int b = p / NHW, rem = p % NHW;
  int y = rem / NW, x = rem % NW;
  const float* wc = w + c * 9;
  float s = bias[c];
#pragma unroll
  for (int t = 0; t < 9; ++t) {
    int ys = y + t / 3 - 1, xs = x + t % 3 - 1;
    if (ys >= 0 && ys < NH && xs >= 0 && xs < NW)
      s += wc[t] * in[(((long)b * NC + c) * NH + ys) * NW + xs];
  }
  out[tid] = (_Float16)s;
}

// ---------------------------------------------------------------------------
// Implicit-GEMM conv via WMMA f32_16x16x32_f16, fully templated.
//   Block = 256 threads = 8 waves = 8 consecutive 16-pixel M-tiles, all
//   sharing one 64-cout group. B (weights) staged into LDS double-buffered,
//   via GLOBAL_LOAD_ASYNC_TO_LDS_B128 when the builtin exists (ASYNCcnt),
//   else global load + ds_store. Each wave runs 4 WMMAs reusing one A frag.
//   MODE 0: out_h NHWC f16 (stride COUT); MODE 1: out_f NCHW f32.
// ---------------------------------------------------------------------------
template<int CINP, int COUT, int TAPS, int LRELU, int MODE>
__global__ __launch_bounds__(256) void k_conv_wmma_t(
    const _Float16* __restrict__ in, const _Float16* __restrict__ wk,
    const float* __restrict__ bias, _Float16* __restrict__ out_h,
    float* __restrict__ out_f) {
  constexpr int NGROUPS = COUT / NGROUP_CO;
  constexpr int KCHUNKS = CINP / 32;
  constexpr int NSTEPS  = TAPS * KCHUNKS;
  constexpr long KLEN   = (long)TAPS * CINP;

  __shared__ _Float16 ldsB[2][NGROUP_CO * 32];   // 2 x 4 KB

  const int tid  = threadIdx.x;
  const int wave = tid >> 5;
  const int lane = tid & 31;
  const int nGroup = (int)(blockIdx.x % NGROUPS);
  const int mBlock = (int)(blockIdx.x / NGROUPS);
  const int mTile  = mBlock * WAVES_PB + wave;
  const int n0     = nGroup * NGROUP_CO;

  const int b  = mTile / (NH * TILES_X);
  const int r  = mTile % (NH * TILES_X);
  const int y  = r / TILES_X;
  const int x0 = (r % TILES_X) << 4;

  const int half = lane >> 4;
  const int mrow = lane & 15;        // A-matrix row for this lane
  const int ncol = lane & 15;        // B/D column for this lane
  const int xA   = x0 + mrow;

  // cooperative B staging: thread -> (co, 8-K chunk)
  const int ldCo = tid >> 2;                 // 0..63
  const int ldK  = (tid & 3) << 3;           // 0,8,16,24
  const _Float16* wRow = wk + (long)(n0 + ldCo) * KLEN + ldK;
  _Float16* dst0 = &ldsB[0][ldCo * 32 + ldK];
  _Float16* dst1 = &ldsB[1][ldCo * 32 + ldK];

  v8f acc[NSUB] = {};

  auto stageTo = [&](int step, _Float16* dst) {
    int t  = step / KCHUNKS;
    int kc = (step % KCHUNKS) * 32;
    const _Float16* src = wRow + (long)t * CINP + kc;
    __builtin_prefetch(src + 64, 0, 3);                    // global_prefetch_b8
#if USE_ASYNC_LDS
    __builtin_amdgcn_global_load_async_to_lds_b128(
        (__attribute__((address_space(1))) v4i*)src,
        (__attribute__((address_space(3))) v4i*)dst, 0, 0);
#else
    v8h tmp = *(const v8h*)src;
    *(v8h*)dst = tmp;
#endif
  };

  // prologue: stage step 0 into buffer 0
  stageTo(0, dst0);

  int cur = 0;
  for (int step = 0; step < NSTEPS; ++step) {
#if USE_ASYNC_LDS
    __builtin_amdgcn_s_wait_asynccnt(0);     // my async stores to LDS landed
#endif
    __syncthreads();                         // buffer `cur` ready, `cur^1` free
    if (step + 1 < NSTEPS) stageTo(step + 1, cur ? dst0 : dst1);

    // A fragment (per-lane global load, border zero-fill)
    const int t  = step / KCHUNKS;
    const int kc = (step % KCHUNKS) * 32;
    const int dy = (TAPS == 1) ? 0 : (t / 3 - 1);
    const int dx = (TAPS == 1) ? 0 : (t % 3 - 1);
    const int ys = y + dy;
    const int xs = xA + dx;
    const bool valid = (ys >= 0) && (ys < NH) && (xs >= 0) && (xs < NW);
    union { v16h h16; v8h h8[2]; } af;
    if (valid) {
      const _Float16* aRow = in + (long)((b * NH + ys) * NW + xs) * CINP;
      af.h8[0] = *(const v8h*)(aRow + kc + half * 16);      // K = kc+h*16   .. +7
      af.h8[1] = *(const v8h*)(aRow + kc + half * 16 + 8);  // K = kc+h*16+8 .. +15
    } else {
      af.h16 = zero_v16h();
    }

    const _Float16* bb = &ldsB[cur][0];
#pragma unroll
    for (int s = 0; s < NSUB; ++s) {
      v16h bf = *(const v16h*)(bb + (s * 16 + ncol) * 32 + half * 16); // ds_load
      acc[s] = __builtin_amdgcn_wmma_f32_16x16x32_f16(
          /*neg_a=*/false, af.h16, /*neg_b=*/false, bf,
          /*c_mod=*/(short)0, acc[s], /*reuse_a=*/false, /*reuse_b=*/false);
    }
    cur ^= 1;
  }

#pragma unroll
  for (int s = 0; s < NSUB; ++s) {
    const int co = n0 + s * 16 + ncol;
    const float bv = bias[co];
    if (MODE == 0) {
#pragma unroll
      for (int ri = 0; ri < 8; ++ri) {
        int m = half * 8 + ri;
        float v = acc[s][ri] + bv;
        if (LRELU) v = (v >= 0.0f) ? v : 0.2f * v;
        out_h[(long)(mTile * 16 + m) * COUT + co] = (_Float16)v;
      }
    } else {
#pragma unroll
      for (int ri = 0; ri < 8; ++ri) {
        int m = half * 8 + ri;
        int x = x0 + m;
        float v = acc[s][ri] + bv;
        if (LRELU) v = (v >= 0.0f) ? v : 0.2f * v;
        out_f[(((long)b * COUT + co) * NH + y) * NW + x] = v;
      }
    }
  }
}

// ---------------------------------------------------------------------------
// 1x1 projection (Cin -> 9) + softmax; whpMode: whp = -sm, whp[4]+=1
// ---------------------------------------------------------------------------
__global__ void k_proj9(const _Float16* __restrict__ in, int Cin,
                        const float* __restrict__ w, const float* __restrict__ bias,
                        float* __restrict__ out9, int whpMode) {
  int p = blockIdx.x * blockDim.x + threadIdx.x;
  if (p >= NPIX) return;
  const _Float16* v = in + (long)p * Cin;
  float s[9];
#pragma unroll
  for (int k = 0; k < 9; ++k) s[k] = bias[k];
  for (int c0 = 0; c0 < Cin; c0 += 8) {
    v8h av = *(const v8h*)(v + c0);
    float af[8];
#pragma unroll
    for (int i = 0; i < 8; ++i) af[i] = (float)av[i];
#pragma unroll
    for (int k = 0; k < 9; ++k) {
      const float* wr = w + (long)k * Cin + c0;
#pragma unroll
      for (int i = 0; i < 8; ++i) s[k] += af[i] * wr[i];
    }
  }
  float mx = s[0];
#pragma unroll
  for (int k = 1; k < 9; ++k) mx = fmaxf(mx, s[k]);
  float den = 0.0f, e[9];
#pragma unroll
  for (int k = 0; k < 9; ++k) { e[k] = expf(s[k] - mx); den += e[k]; }
  float inv = 1.0f / den;
#pragma unroll
  for (int k = 0; k < 9; ++k) {
    float sm = e[k] * inv;
    out9[(long)p * 9 + k] = whpMode ? ((k == 4) ? 1.0f - sm : -sm) : sm;
  }
}

// ---------------------------------------------------------------------------
// CARAFE smoothing: f_smooth[c,p] = sum_tap dec[c, p+tap]*kw[p,tap]
// writes channels [0..255] of feat (stride 288)
// ---------------------------------------------------------------------------
__global__ void k_carafe(const _Float16* __restrict__ dec, const float* __restrict__ kw9,
                         _Float16* __restrict__ feat) {
  long tid = (long)blockIdx.x * blockDim.x + threadIdx.x;
  if (tid >= (long)NPIX * 32) return;
  int p  = (int)(tid >> 5);
  int c0 = (int)(tid & 31) << 3;
  int b = p / NHW, rem = p % NHW;
  int y = rem / NW, x = rem % NW;
  float kw[9];
#pragma unroll
  for (int t = 0; t < 9; ++t) kw[t] = kw9[(long)p * 9 + t];
  float acc[8] = {0, 0, 0, 0, 0, 0, 0, 0};
#pragma unroll
  for (int t = 0; t < 9; ++t) {
    int ys = y + t / 3 - 1, xs = x + t % 3 - 1;
    if (ys >= 0 && ys < NH && xs >= 0 && xs < NW) {
      v8h nb = *(const v8h*)(dec + (long)((b * NH + ys) * NW + xs) * NC + c0);
#pragma unroll
      for (int i = 0; i < 8; ++i) acc[i] += kw[t] * (float)nb[i];
    }
  }
#pragma unroll
  for (int i = 0; i < 8; ++i) feat[(long)p * 288 + c0 + i] = (_Float16)acc[i];
}

// ---------------------------------------------------------------------------
// Cosine similarity of f_smooth with its 3x3 neighborhood -> feat ch[256..264],
// zero-fill pad channels [265..287]. Channel loop outermost, v8h loads, all
// 9 tap accumulators updated per chunk (center read exactly once).
// ---------------------------------------------------------------------------
__global__ void k_sim(_Float16* __restrict__ feat) {
  int p = blockIdx.x * blockDim.x + threadIdx.x;
  if (p >= NPIX) return;
  int b = p / NHW, rem = p % NHW;
  int y = rem / NW, x = rem % NW;
  const _Float16* ctr = feat + (long)p * 288;
  const _Float16* nbp[9];
  bool val[9];
#pragma unroll
  for (int t = 0; t < 9; ++t) {
    int ys = y + t / 3 - 1, xs = x + t % 3 - 1;
    val[t] = (ys >= 0 && ys < NH && xs >= 0 && xs < NW);
    nbp[t] = val[t] ? (feat + (long)((b * NH + ys) * NW + xs) * 288) : ctr;
  }
  float cs = 0.0f, num[9], ns[9];
#pragma unroll
  for (int t = 0; t < 9; ++t) { num[t] = 0.0f; ns[t] = 0.0f; }
  for (int c0 = 0; c0 < NC; c0 += 8) {
    v8h cv = *(const v8h*)(ctr + c0);
    float cf[8];
#pragma unroll
    for (int i = 0; i < 8; ++i) { cf[i] = (float)cv[i]; cs += cf[i] * cf[i]; }
#pragma unroll
    for (int t = 0; t < 9; ++t) {
      v8h nv = *(const v8h*)(nbp[t] + c0);
#pragma unroll
      for (int i = 0; i < 8; ++i) {
        float nf = (float)nv[i];
        num[t] += cf[i] * nf;
        ns[t]  += nf * nf;
      }
    }
  }
  float cn = sqrtf(cs) + 1e-8f;
#pragma unroll
  for (int t = 0; t < 9; ++t) {
    float sim = val[t] ? (num[t] / (cn * (sqrtf(ns[t]) + 1e-8f))) : 0.0f;
    feat[(long)p * 288 + 256 + t] = (_Float16)sim;
  }
#pragma unroll
  for (int c = 265; c < 288; ++c) feat[(long)p * 288 + c] = (_Float16)0.0f;
}

// ---------------------------------------------------------------------------
// Offset head: 3x3 conv 64->2 + tanh*0.1 (v8h activation loads)
// ---------------------------------------------------------------------------
__global__ void k_offhead(const _Float16* __restrict__ t2, const float* __restrict__ w,
                          const float* __restrict__ bias, float* __restrict__ off) {
  int p = blockIdx.x * blockDim.x + threadIdx.x;
  if (p >= NPIX) return;
  int b = p / NHW, rem = p % NHW;
  int y = rem / NW, x = rem % NW;
  float a0 = bias[0], a1 = bias[1];
#pragma unroll
  for (int t = 0; t < 9; ++t) {
    int ys = y + t / 3 - 1, xs = x + t % 3 - 1;
    if (ys < 0 || ys >= NH || xs < 0 || xs >= NW) continue;
    const _Float16* nb = t2 + (long)((b * NH + ys) * NW + xs) * 64;
    for (int c0 = 0; c0 < 64; c0 += 8) {
      v8h nv = *(const v8h*)(nb + c0);
#pragma unroll
      for (int i = 0; i < 8; ++i) {
        float v = (float)nv[i];
        a0 += v * w[(long)(c0 + i) * 9 + t];
        a1 += v * w[(long)(64 + c0 + i) * 9 + t];
      }
    }
  }
  off[(long)p * 2 + 0] = tanhf(a0) * 0.1f;
  off[(long)p * 2 + 1] = tanhf(a1) * 0.1f;
}

// ---------------------------------------------------------------------------
// f_out = grid_sample_border(f_smooth, gx, gy) + enc + sum_tap enc[nbr]*whp
// feat ch[0..255] == f_smooth (stride 288); output fout NHWC f16 (stride 256)
// ---------------------------------------------------------------------------
__global__ void k_fuse(const _Float16* __restrict__ feat, const _Float16* __restrict__ enc,
                       const float* __restrict__ whp9, const float* __restrict__ off,
                       _Float16* __restrict__ fout) {
  long tid = (long)blockIdx.x * blockDim.x + threadIdx.x;
  if (tid >= (long)NPIX * 32) return;
  int p  = (int)(tid >> 5);
  int c0 = (int)(tid & 31) << 3;
  int b = p / NHW, rem = p % NHW;
  int y = rem / NW, x = rem % NW;

  float gx = -1.0f + 2.0f * x / (float)(NW - 1) + off[(long)p * 2 + 0];
  float gy = -1.0f + 2.0f * y / (float)(NH - 1) + off[(long)p * 2 + 1];
  float px = fminf(fmaxf((gx + 1.0f) * 0.5f * (NW - 1), 0.0f), (float)(NW - 1));
  float py = fminf(fmaxf((gy + 1.0f) * 0.5f * (NH - 1), 0.0f), (float)(NH - 1));
  float x0f = floorf(px), y0f = floorf(py);
  float wx = px - x0f, wy = py - y0f;
  int x0i = (int)x0f, y0i = (int)y0f;
  int x1i = min(x0i + 1, NW - 1), y1i = min(y0i + 1, NH - 1);
  long base = (long)b * NHW;
  const _Float16* s00 = feat + (base + (long)y0i * NW + x0i) * 288 + c0;
  const _Float16* s01 = feat + (base + (long)y0i * NW + x1i) * 288 + c0;
  const _Float16* s10 = feat + (base + (long)y1i * NW + x0i) * 288 + c0;
  const _Float16* s11 = feat + (base + (long)y1i * NW + x1i) * 288 + c0;
  float w00 = (1.0f - wx) * (1.0f - wy), w01 = wx * (1.0f - wy);
  float w10 = (1.0f - wx) * wy,          w11 = wx * wy;

  float kw[9];
#pragma unroll
  for (int t = 0; t < 9; ++t) kw[t] = whp9[(long)p * 9 + t];

  float acc[8];
  v8h a00 = *(const v8h*)s00, a01 = *(const v8h*)s01;
  v8h a10 = *(const v8h*)s10, a11 = *(const v8h*)s11;
  v8h ec = *(const v8h*)(enc + (long)p * NC + c0);
#pragma unroll
  for (int i = 0; i < 8; ++i)
    acc[i] = w00 * (float)a00[i] + w01 * (float)a01[i] +
             w10 * (float)a10[i] + w11 * (float)a11[i] + (float)ec[i];
#pragma unroll
  for (int t = 0; t < 9; ++t) {
    int ys = y + t / 3 - 1, xs = x + t % 3 - 1;
    if (ys >= 0 && ys < NH && xs >= 0 && xs < NW) {
      v8h nb = *(const v8h*)(enc + (long)((b * NH + ys) * NW + xs) * NC + c0);
#pragma unroll
      for (int i = 0; i < 8; ++i) acc[i] += kw[t] * (float)nb[i];
    }
  }
#pragma unroll
  for (int i = 0; i < 8; ++i) fout[(long)p * NC + c0 + i] = (_Float16)acc[i];
}

// ---------------------------------------------------------------------------
// Host-side orchestration
// ---------------------------------------------------------------------------
extern "C" void kernel_launch(void* const* d_in, const int* in_sizes, int n_in,
                              void* d_out, int out_size, void* d_ws, size_t ws_size,
                              hipStream_t stream) {
  const float* dec       = (const float*)d_in[0];
  const float* enc       = (const float*)d_in[1];
  const float* alpf_dw_w = (const float*)d_in[2];
  const float* alpf_dw_b = (const float*)d_in[3];
  const float* alpf_c1_w = (const float*)d_in[4];
  const float* alpf_c1_b = (const float*)d_in[5];
  const float* alpf_c2_w = (const float*)d_in[6];
  const float* alpf_c2_b = (const float*)d_in[7];
  const float* ahpf_c1_w = (const float*)d_in[8];
  const float* ahpf_c1_b = (const float*)d_in[9];
  const float* ahpf_c2_w = (const float*)d_in[10];
  const float* ahpf_c2_b = (const float*)d_in[11];
  const float* off_c1_w  = (const float*)d_in[12];
  const float* off_c1_b  = (const float*)d_in[13];
  const float* off_c2_w  = (const float*)d_in[14];
  const float* off_c2_b  = (const float*)d_in[15];
  const float* fus_w     = (const float*)d_in[16];
  const float* fus_b     = (const float*)d_in[17];
  float* outp = (float*)d_out;

  char* ws = (char*)d_ws;
  size_t cur = 0;
  auto alloc = [&](size_t bytes) -> char* {
    cur = (cur + 255) & ~(size_t)255;
    char* p = ws + cur; cur += bytes; return p;
  };

  _Float16* dec_h  = (_Float16*)alloc((size_t)NPIX * 256 * 2);
  _Float16* enc_h  = (_Float16*)alloc((size_t)NPIX * 256 * 2);
  _Float16* kw0    = (_Float16*)alloc((size_t)NPIX * 256 * 2);   // depthwise out
  _Float16* t1     = (_Float16*)alloc((size_t)NPIX * 128 * 2);   // alpf_c1 out
  float*    kw9    = (float*)   alloc((size_t)NPIX * 9 * 4);     // softmax weights
  _Float16* feat   = (_Float16*)alloc((size_t)NPIX * 288 * 2);   // f_smooth(256)+sim(9)+pad
  _Float16* t2     = (_Float16*)alloc((size_t)NPIX * 64 * 2);    // off_c1 out
  float*    off    = (float*)   alloc((size_t)NPIX * 2 * 4);
  _Float16* t3     = (_Float16*)alloc((size_t)NPIX * 128 * 2);   // ahpf_c1 out
  float*    whp9   = (float*)   alloc((size_t)NPIX * 9 * 4);
  _Float16* fout_h = (_Float16*)alloc((size_t)NPIX * 256 * 2);
  _Float16* wp_a1  = (_Float16*)alloc((size_t)128 * 1 * 256 * 2);
  _Float16* wp_o1  = (_Float16*)alloc((size_t)64  * 9 * 288 * 2);
  _Float16* wp_h1  = (_Float16*)alloc((size_t)128 * 9 * 256 * 2);
  _Float16* wp_fu  = (_Float16*)alloc((size_t)256 * 9 * 256 * 2);

  auto blocks = [](long n) { return (unsigned)((n + 255) / 256); };

  // Weight repacks
  k_repack_w<<<blocks((long)128 * 1 * 256), 256, 0, stream>>>(alpf_c1_w, wp_a1, 128, 256, 256, 1);
  k_repack_w<<<blocks((long)64  * 9 * 288), 256, 0, stream>>>(off_c1_w,  wp_o1, 64,  265, 288, 9);
  k_repack_w<<<blocks((long)128 * 9 * 256), 256, 0, stream>>>(ahpf_c1_w, wp_h1, 128, 256, 256, 9);
  k_repack_w<<<blocks((long)256 * 9 * 256), 256, 0, stream>>>(fus_w,     wp_fu, 256, 256, 256, 9);

  // Layout conversions
  k_nchw2nhwc_h<<<blocks((long)NPIX * 256), 256, 0, stream>>>(dec, dec_h, 256);
  k_nchw2nhwc_h<<<blocks((long)NPIX * 256), 256, 0, stream>>>(enc, enc_h, 256);

  // alpf branch: depthwise -> 1x1(WMMA)+lrelu -> 128->9 softmax -> CARAFE -> sim
  k_dw3x3<<<blocks((long)NPIX * 256), 256, 0, stream>>>(dec, alpf_dw_w, alpf_dw_b, kw0);

  // conv grids: (MTILES/8) m-blocks x (COUT/64) n-groups, 256 threads
  k_conv_wmma_t<256, 128, 1, 1, 0><<<(MTILES / 8) * (128 / 64), 256, 0, stream>>>(
      kw0, wp_a1, alpf_c1_b, t1, nullptr);

  k_proj9<<<blocks(NPIX), 256, 0, stream>>>(t1, 128, alpf_c2_w, alpf_c2_b, kw9, 0);
  k_carafe<<<blocks((long)NPIX * 32), 256, 0, stream>>>(dec_h, kw9, feat);
  k_sim<<<blocks(NPIX), 256, 0, stream>>>(feat);

  // offset branch: 3x3 288->64 (WMMA)+lrelu -> 3x3 64->2 tanh*0.1
  k_conv_wmma_t<288, 64, 9, 1, 0><<<(MTILES / 8) * (64 / 64), 256, 0, stream>>>(
      feat, wp_o1, off_c1_b, t2, nullptr);
  k_offhead<<<blocks(NPIX), 256, 0, stream>>>(t2, off_c2_w, off_c2_b, off);

  // ahpf branch: 3x3 256->128 (WMMA)+lrelu -> 128->9 softmax -> whp transform
  k_conv_wmma_t<256, 128, 9, 1, 0><<<(MTILES / 8) * (128 / 64), 256, 0, stream>>>(
      enc_h, wp_h1, ahpf_c1_b, t3, nullptr);
  k_proj9<<<blocks(NPIX), 256, 0, stream>>>(t3, 128, ahpf_c2_w, ahpf_c2_b, whp9, 1);

  // resample + high-pass fuse
  k_fuse<<<blocks((long)NPIX * 32), 256, 0, stream>>>(feat, enc_h, whp9, off, fout_h);

  // final 3x3 256->256 (WMMA) -> NCHW f32 output
  k_conv_wmma_t<256, 256, 9, 0, 1><<<(MTILES / 8) * (256 / 64), 256, 0, stream>>>(
      fout_h, wp_fu, fus_b, nullptr, outp);
}
